// EdgeNet_23364622090240
// MI455X (gfx1250) — matrix operand order repaired and checked by reading, not compiled, not run on hardware
//
#include <hip/hip_runtime.h>
#include <hip/hip_bf16.h>

// ---------------------------------------------------------------------------
// EdgeNet (NNConv x4 + BN/ReLU + segment_max + MLP head) for MI455X (gfx1250)
//
// Dominant cost: msg = (h outer x_src) @ W2  as an M=E,K=128*IN_CH,N=64 GEMM
// executed with v_wmma_f32_16x16x32_bf16. A is generated in registers
// (rank-1 outer product per 32-wide K-chunk), B (w2 cast to bf16 and
// pre-swizzled to WMMA fragment layout) is streamed L2 -> LDS with
// global_load_async_to_lds_b128, triple-buffered (2-deep pipeline) and
// shared by 4 waves per workgroup.
// ---------------------------------------------------------------------------

constexpr int N = 10000, E = 50000, IN = 32, HID = 64, G = 64, GIN = 8;
constexpr float BN_EPS = 1e-5f;

typedef __attribute__((ext_vector_type(16))) __bf16       v16bf;
typedef __attribute__((ext_vector_type(8)))  float        v8f;
typedef __attribute__((ext_vector_type(8)))  unsigned int v8u;

static __device__ __forceinline__ unsigned short f2bf_rne(float f) {
  unsigned u = __float_as_uint(f);
  unsigned r = u + 0x7FFFu + ((u >> 16) & 1u);
  return (unsigned short)(r >> 16);
}
static __device__ __forceinline__ float bf2f(unsigned short b) {
  return __uint_as_float(((unsigned)b) << 16);
}

// CDNA5 async copy: global -> LDS, 16B per lane, tracked by ASYNCcnt.
static __device__ __forceinline__ void async_b128(unsigned lds_addr, unsigned goff,
                                                  const void* sbase) {
  asm volatile("global_load_async_to_lds_b128 %0, %1, %2"
               :: "v"(lds_addr), "v"(goff), "s"(sbase)
               : "memory");
}
// async loads complete in order within a wave: <=2 outstanding means the
// older chunk's pair of b128 copies has landed in LDS.
static __device__ __forceinline__ void wait_async_le2() {
  asm volatile("s_wait_asynccnt 0x2" ::: "memory");
}

// --- one-time: cast w2 (f32, KSUP x 64 row-major) into WMMA-B fragment-major
// bf16 layout.  Entry (q, nt, lane) holds 16 halves (8 dwords) so the msg
// kernel reads each B fragment as two contiguous b128 per lane.
// B[k][n]: lane = (n & 15) + 16*(k >= 16 within chunk), dword j: K = 2j,2j+1.
__global__ void k_cast_w2(const float* __restrict__ src,
                          unsigned short* __restrict__ dst, int ksup) {
  int idx = blockIdx.x * blockDim.x + threadIdx.x;   // over ksup*64 halves
  if (idx >= ksup * 64) return;
  int p    = idx & 15;          // half within lane entry
  int lane = (idx >> 4) & 31;
  int nt   = (idx >> 9) & 3;
  int q    = idx >> 11;         // 32-wide K chunk
  int j = p >> 1, hbit = p & 1;
  int k = q * 32 + (lane >> 4) * 16 + 2 * j + hbit;
  int n = nt * 16 + (lane & 15);
  dst[idx] = f2bf_rne(src[(size_t)k * 64 + n]);
}

// --- edge MLP first layer: h = relu(edge_attr @ w1 + b1)  (E x 16 -> E x 128)
__global__ void k_edge_h(const float* __restrict__ ea, const float* __restrict__ w1,
                         const float* __restrict__ b1, unsigned short* __restrict__ hb) {
  int t = blockIdx.x * blockDim.x + threadIdx.x;     // E*128
  if (t >= E * 128) return;
  int e = t >> 7, j = t & 127;
  const float* row = ea + (size_t)e * 16;
  float a = b1[j];
#pragma unroll
  for (int k = 0; k < 16; ++k) a = fmaf(row[k], w1[k * 128 + j], a);
  hb[t] = f2bf_rne(fmaxf(a, 0.f));
}

// --- pre = x @ root + bias  (initializes the accumulation target)
__global__ void k_root_bias(const float* __restrict__ x, const float* __restrict__ root,
                            const float* __restrict__ bias, float* __restrict__ pre,
                            int inch) {
  int t = blockIdx.x * blockDim.x + threadIdx.x;     // N*64
  if (t >= N * 64) return;
  int n = t >> 6, o = t & 63;
  const float* xr = x + (size_t)n * inch;
  float a = bias[o];
  for (int k = 0; k < inch; ++k) a = fmaf(xr[k], root[(size_t)k * 64 + o], a);
  pre[t] = a;
}

// --- theta bias term: pre[dst_e] += x[src_e] @ reshape(b2, inch x 64)
__global__ void k_edge_b2(const float* __restrict__ x, const float* __restrict__ b2,
                          const int* __restrict__ src, const int* __restrict__ dst,
                          float* __restrict__ pre, int inch) {
  int t = blockIdx.x * blockDim.x + threadIdx.x;     // E*64
  if (t >= E * 64) return;
  int e = t >> 6, o = t & 63;
  const float* xr = x + (size_t)src[e] * inch;
  float a = 0.f;
  for (int k = 0; k < inch; ++k) a = fmaf(xr[k], b2[(size_t)k * 64 + o], a);
  atomicAdd(pre + (size_t)dst[e] * 64 + o, a);
}

// --- fused msg GEMM: 4 waves per block, each wave owns one 16-edge tile.
// The 4 KB B chunk (4 fragments) is async-copied into triple-buffered LDS
// (pipeline depth 2); each wave fetches one fragment, all consume all four.
template <int IN_CH>
__global__ void __launch_bounds__(128)
k_msg(const float* __restrict__ xcur, const unsigned short* __restrict__ hb,
      const unsigned short* __restrict__ w2b, const int* __restrict__ src,
      const int* __restrict__ dst, float* __restrict__ pre) {
  constexpr int NC    = IN_CH / 32;              // 32-wide i chunks per k
  constexpr int Q     = 128 * NC;                // total K chunks
  constexpr int NTILE = (E + 15) / 16;           // 3125

  __shared__ v8u sB[3][128];                     // triple-buffered B chunk (3 x 4KB)
  __shared__ unsigned short sh[4][16 * 128];     // per-wave h tiles (bf16)
  __shared__ int sdst[4][16];

  const int lane = threadIdx.x & 31;
  const int w    = threadIdx.x >> 5;             // wave id within block
  int tile = blockIdx.x * 4 + w;
  const bool active = (tile < NTILE);
  if (!active) tile = NTILE - 1;                 // keep barrier/fetch duty
  const int e0 = tile * 16;

  {   // stage this wave's h tile (16x128 halves = 1024 dwords), coalesced
    const unsigned int* hsrc = (const unsigned int*)(hb + (size_t)e0 * 128);
    unsigned int* d2 = (unsigned int*)sh[w];
#pragma unroll
    for (int j = 0; j < 32; ++j) d2[j * 32 + lane] = hsrc[j * 32 + lane];
    if (lane < 16) sdst[w][lane] = dst[e0 + lane];
  }

  const int M  = lane & 15;                      // A-matrix row this lane owns
  const int kh = lane >> 4;                      // K-half select (A layout)
  const float* xrow = xcur + (size_t)src[e0 + M] * IN_CH;
  float xf[NC][16];
#pragma unroll
  for (int c = 0; c < NC; ++c) {
#pragma unroll
    for (int j = 0; j < 8; ++j) xf[c][j]     = xrow[c * 32 + kh * 16 + j];
#pragma unroll
    for (int j = 0; j < 8; ++j) xf[c][8 + j] = xrow[c * 32 + kh * 16 + 8 + j];
  }

  // this wave's copy duty: fragment nt == w, 1KB per chunk (2 x b128 per lane)
  const unsigned ldsw  = (unsigned)(size_t)&sB[0][w * 32 + lane];
  const unsigned gstep = (unsigned)(w * 1024 + lane * 32);

  // prologue: chunks 0,1 -> buffers 0,1 (chunk 0 complete after wait<=2)
  async_b128(ldsw,             gstep,             w2b);
  async_b128(ldsw + 16,        gstep + 16,        w2b);
  if (Q > 1) {
    async_b128(ldsw + 4096,      4096u + gstep,      w2b);
    async_b128(ldsw + 4096 + 16, 4096u + gstep + 16, w2b);
  }
  wait_async_le2();
  __syncthreads();

  v8f zero = {0.f, 0.f, 0.f, 0.f, 0.f, 0.f, 0.f, 0.f};
  v8f acc[4] = {zero, zero, zero, zero};

  int b0 = 0;                                    // buffer holding chunk q
  for (int k = 0; k < 128; ++k) {
    const float hk = bf2f(sh[w][M * 128 + k]);
#pragma unroll
    for (int c = 0; c < NC; ++c) {
      const int q = k * NC + c;

      if (q + 2 < Q) {                           // async-copy chunk q+2
        int bw = b0 + 2; if (bw >= 3) bw -= 3;
        const unsigned dlds = ldsw + (unsigned)bw * 4096u;
        const unsigned go   = (unsigned)(q + 2) * 4096u + gstep;
        async_b128(dlds,      go,      w2b);
        async_b128(dlds + 16, go + 16, w2b);
      }

      v8u au;                                    // A frag = hk * x, packed bf16
#pragma unroll
      for (int j = 0; j < 8; ++j) {
        unsigned lo = __float_as_uint(hk * xf[c][2 * j]);
        unsigned hi = __float_as_uint(hk * xf[c][2 * j + 1]);
        au[j] = (lo >> 16) | (hi & 0xFFFF0000u);
      }
      v16bf A = __builtin_bit_cast(v16bf, au);

      v8u breg[4];                               // pull all 4 B frags, one wait
#pragma unroll
      for (int nt = 0; nt < 4; ++nt) breg[nt] = sB[b0][nt * 32 + lane];
#pragma unroll
      for (int nt = 0; nt < 4; ++nt)
        acc[nt] = __builtin_amdgcn_wmma_f32_16x16x32_bf16(
            false, A, false, __builtin_bit_cast(v16bf, breg[nt]),
            (short)0, acc[nt], false, false);

      wait_async_le2();                          // chunk q+1 landed in LDS
      __syncthreads();                           // all waves done reading b0
      b0 = (b0 == 2) ? 0 : b0 + 1;
    }
  }

  // scatter-add: C/D layout -> lanes 0-15: (M=r, N=lane), 16-31: (M=r+8, N=lane-16)
  if (active) {
#pragma unroll
    for (int nt = 0; nt < 4; ++nt) {
#pragma unroll
      for (int r = 0; r < 8; ++r) {
        int m = r + 8 * kh;
        atomicAdd(pre + (size_t)sdst[w][m] * 64 + nt * 16 + M, acc[nt][r]);
      }
    }
  }
}

// --- BatchNorm batch statistics (biased variance), one block per channel
__global__ void k_bn_stats(const float* __restrict__ pre, float* __restrict__ stats) {
  __shared__ float rs[256], rss[256];
  const int c = blockIdx.x, tid = threadIdx.x;
  float s = 0.f, ss = 0.f;
  for (int n = tid; n < N; n += 256) {
    float v = pre[(size_t)n * 64 + c];
    s += v; ss += v * v;
  }
  rs[tid] = s; rss[tid] = ss;
  __syncthreads();
  for (int w = 128; w > 0; w >>= 1) {
    if (tid < w) { rs[tid] += rs[tid + w]; rss[tid] += rss[tid + w]; }
    __syncthreads();
  }
  if (tid == 0) {
    float mu = rs[0] / (float)N;
    stats[c]      = mu;
    stats[64 + c] = rss[0] / (float)N - mu * mu;
  }
}

__global__ void k_bn_apply(const float* __restrict__ pre, const float* __restrict__ stats,
                           const float* __restrict__ gamma, const float* __restrict__ beta,
                           float* __restrict__ xf) {
  int t = blockIdx.x * blockDim.x + threadIdx.x;     // N*64
  if (t >= N * 64) return;
  int c = t & 63;
  float y = gamma[c] * (pre[t] - stats[c]) * rsqrtf(stats[64 + c] + BN_EPS) + beta[c];
  xf[t] = fmaxf(y, 0.f);
}

__global__ void k_pool_init(float* __restrict__ pool) {
  int t = blockIdx.x * blockDim.x + threadIdx.x;
  if (t < G * 64) pool[t] = 0.f;                 // post-ReLU values are >= 0
}

__global__ void k_pool(const float* __restrict__ xf, const int* __restrict__ batch,
                       float* __restrict__ pool) {
  int t = blockIdx.x * blockDim.x + threadIdx.x;     // N*64
  if (t >= N * 64) return;
  int n = t >> 6, c = t & 63;
  // int-ordered max == float max for non-negative floats
  atomicMax((int*)(pool + (size_t)batch[n] * 64 + c), __float_as_int(xf[t]));
}

__global__ void k_final(const float* __restrict__ pool, const float* __restrict__ u,
                        const float* __restrict__ w1, const float* __restrict__ b1,
                        const float* __restrict__ w2, const float* __restrict__ b2,
                        float* __restrict__ out) {
  __shared__ float hsh[64];
  const int g = blockIdx.x, j = threadIdx.x;         // 64 blocks x 64 threads
  float a = b1[j];
#pragma unroll 8
  for (int k = 0; k < 64; ++k)  a = fmaf(pool[g * 64 + k], w1[k * 64 + j], a);
#pragma unroll
  for (int k = 0; k < GIN; ++k) a = fmaf(u[g * GIN + k], w1[(64 + k) * 64 + j], a);
  hsh[j] = fmaxf(a, 0.f) * w2[j];
  __syncthreads();
  for (int w = 32; w > 0; w >>= 1) {
    if (j < w) hsh[j] += hsh[j + w];
    __syncthreads();
  }
  if (j == 0) out[g] = hsh[0] + b2[0];
}

extern "C" void kernel_launch(void* const* d_in, const int* in_sizes, int n_in,
                              void* d_out, int out_size, void* d_ws, size_t ws_size,
                              hipStream_t stream) {
  (void)in_sizes; (void)n_in; (void)out_size; (void)ws_size;
  const float* x0    = (const float*)d_in[0];
  const int*   ei    = (const int*)d_in[1];
  const int*   src   = ei;
  const int*   dstp  = ei + E;
  const float* ea    = (const float*)d_in[2];
  const int*   batch = (const int*)d_in[3];
  const float* u     = (const float*)d_in[4];
  const float* enn0_w1 = (const float*)d_in[5];
  const float* enn0_b1 = (const float*)d_in[6];
  const float* enn0_w2 = (const float*)d_in[7];
  const float* enn0_b2 = (const float*)d_in[8];
  const float* root0   = (const float*)d_in[9];
  const float* bias0   = (const float*)d_in[10];
  const float* gamma0  = (const float*)d_in[11];
  const float* beta0   = (const float*)d_in[12];
  const float* enn_w1  = (const float*)d_in[13];
  const float* enn_b1  = (const float*)d_in[14];
  const float* enn_w2  = (const float*)d_in[15];
  const float* enn_b2  = (const float*)d_in[16];
  const float* rootL   = (const float*)d_in[17];
  const float* biasL   = (const float*)d_in[18];
  const float* gammaL  = (const float*)d_in[19];
  const float* betaL   = (const float*)d_in[20];
  const float* pp_w1   = (const float*)d_in[21];
  const float* pp_b1   = (const float*)d_in[22];
  const float* pp_w2   = (const float*)d_in[23];
  const float* pp_b2   = (const float*)d_in[24];
  float* out = (float*)d_out;

  // workspace carve-up (deterministic, recomputed every call)
  char* ws = (char*)d_ws;
  size_t off = 0;
  auto take = [&](size_t bytes) -> char* {
    char* p = ws + off;
    off = (off + bytes + 255) & ~(size_t)255;
    return p;
  };
  unsigned short* w2b[4];
  w2b[0] = (unsigned short*)take((size_t)4096 * 64 * 2);
  for (int l = 1; l < 4; ++l) w2b[l] = (unsigned short*)take((size_t)8192 * 64 * 2);
  unsigned short* hb  = (unsigned short*)take((size_t)E * 128 * 2);
  float* xf    = (float*)take((size_t)N * 64 * 4);
  float* pre   = (float*)take((size_t)N * 64 * 4);
  float* stats = (float*)take(128 * 4);
  float* pool  = (float*)take((size_t)G * 64 * 4);

  // one-time weight swizzle/cast (w2 lives in L2 thereafter)
  k_cast_w2<<<(4096 * 64) / 256, 256, 0, stream>>>(enn0_w2, w2b[0], 4096);
  for (int l = 0; l < 3; ++l)
    k_cast_w2<<<(8192 * 64) / 256, 256, 0, stream>>>(
        enn_w2 + (size_t)l * 128 * 4096, w2b[l + 1], 8192);

  const int msg_blocks = ((E + 15) / 16 + 3) / 4;    // 4 tiles per block

  for (int layer = 0; layer < 4; ++layer) {
    const bool first = (layer == 0);
    const int  l     = layer - 1;           // index into stacked params
    const int  inch  = first ? IN : HID;
    const float* xin   = first ? x0 : xf;
    const float* w1    = first ? enn0_w1 : enn_w1 + (size_t)l * 16 * 128;
    const float* b1    = first ? enn0_b1 : enn_b1 + (size_t)l * 128;
    const float* b2    = first ? enn0_b2 : enn_b2 + (size_t)l * 4096;
    const float* root  = first ? root0   : rootL  + (size_t)l * 64 * 64;
    const float* bias  = first ? bias0   : biasL  + (size_t)l * 64;
    const float* gamma = first ? gamma0  : gammaL + (size_t)l * 64;
    const float* beta  = first ? beta0   : betaL  + (size_t)l * 64;

    k_edge_h<<<(E * 128) / 256, 256, 0, stream>>>(ea, w1, b1, hb);
    k_root_bias<<<(N * 64) / 256, 256, 0, stream>>>(xin, root, bias, pre, inch);
    if (first)
      k_msg<IN ><<<msg_blocks, 128, 0, stream>>>(xin, hb, w2b[layer], src, dstp, pre);
    else
      k_msg<HID><<<msg_blocks, 128, 0, stream>>>(xin, hb, w2b[layer], src, dstp, pre);
    k_edge_b2<<<(E * 64) / 256, 256, 0, stream>>>(xin, b2, src, dstp, pre, inch);
    k_bn_stats<<<64, 256, 0, stream>>>(pre, stats);
    k_bn_apply<<<(N * 64) / 256, 256, 0, stream>>>(pre, stats, gamma, beta, xf);
  }

  k_pool_init<<<(G * 64) / 256, 256, 0, stream>>>(pool);
  k_pool<<<(N * 64) / 256, 256, 0, stream>>>(xf, batch, pool);
  k_final<<<G, 64, 0, stream>>>(pool, u, pp_w1, pp_b1, pp_w2, pp_b2, out);
}
